// CharLevelEncoder_65953517797815
// MI455X (gfx1250) — matrix operand order repaired and checked by reading, not compiled, not run on hardware
//
#include <hip/hip_runtime.h>
#include <hip/hip_bf16.h>
#include <stdint.h>

// ---------------------------------------------------------------------------
// Char-level LSTM encoder for MI455X (gfx1250), bf16 WMMA + f32 accumulate.
// Compute-bound (~350 GFLOP vs ~70MB traffic -> ~5000 FLOP/B).
//  * W_ih/W_hh WMMA B-fragments register-resident across all 16 timesteps
//    (~460 VGPRs/wave of the 1024 wave32 budget; verified no scratch spills).
//  * Gate-aligned wave->column mapping: wave w owns gates (i,f,g,o) of hidden
//    units [16w,16w+16), so the whole LSTM cell (bias, sigmoid/tanh, c-state,
//    length-mask freeze) runs in registers directly on the C fragments.
//  * Only h (bf16, 16KB) round-trips LDS per step; branch-free transcendentals
//    (v_exp_f32 TRANS ops co-execute with 16-bit WMMA).
//  * All staging traffic vectorized to b128.
// ---------------------------------------------------------------------------

typedef __attribute__((ext_vector_type(16))) __bf16 v16bf;
typedef __attribute__((ext_vector_type(8)))  float  v8f;

#define ROWS   32    // words per workgroup (2 M-tiles)
#define HID    256
#define G4     1024
#define CDIM   64
#define TSTEPS 16
#define WDIM   256
#define COMB   512   // WORD_DIM + HIDDEN

__device__ __forceinline__ __bf16 f2bf(float f) {
  union { float f; uint32_t u; } x; x.f = f;
  uint32_t u = x.u;
  u += 0x7FFFu + ((u >> 16) & 1u);          // round-to-nearest-even
  union { unsigned short s; __bf16 b; } y;
  y.s = (unsigned short)(u >> 16);
  return y.b;
}

__device__ __forceinline__ uint32_t pk2bf(float a, float b) {
  union { __bf16 h[2]; uint32_t u; } z;
  z.h[0] = f2bf(a); z.h[1] = f2bf(b);
  return z.u;
}

__device__ __forceinline__ v8f wmma_bf16(v16bf a, v16bf b, v8f c) {
  return __builtin_amdgcn_wmma_f32_16x16x32_bf16(false, a, false, b,
                                                 (short)0, c, false, false);
}

union AFrag { v16bf v; uint4 q[2]; };

// A 16x32 fragment, contiguous k order (B loads use the same): lane<16 holds K[0..15] of row=lane;
// lane>=16 holds K[16..31] of row=lane-16.  Two b128 loads.
__device__ __forceinline__ v16bf load_afrag(const __bf16* buf, int rowElems,
                                            int kt, int lane) {
  const int row  = lane & 15;
  const int half = (lane >> 4) * 16;
  const __bf16* p = buf + row * rowElems + kt * 32 + half;
  AFrag u;
  u.q[0] = *(const uint4*)(p);
  u.q[1] = *(const uint4*)(p + 8);
  return u.v;
}

// Branch-free sigmoid/tanh on v_exp_f32 (TRANS, co-executes with WMMA).
__device__ __forceinline__ float sigmoidf_(float x) {
  return 1.0f / (1.0f + __expf(-x));
}
__device__ __forceinline__ float tanhf_(float x) {
  const float s = 1.0f / (1.0f + __expf(-2.0f * x));  // sigmoid(2x)
  return fmaf(2.0f, s, -1.0f);                        // saturates at +-1
}

// --------------------------- weight pre-pack -------------------------------
__global__ void clenc_pack_kernel(const float* __restrict__ wih,
                                  const float* __restrict__ whh,
                                  const float* __restrict__ wlin,
                                  const float* __restrict__ bih,
                                  const float* __restrict__ bhh,
                                  __bf16* __restrict__ wihb,
                                  __bf16* __restrict__ whhb,
                                  __bf16* __restrict__ wlinb,
                                  float* __restrict__ bcomb) {
  const int i = blockIdx.x * blockDim.x + threadIdx.x;
  const int stride = gridDim.x * blockDim.x;
  for (int k = i; k < G4 * HID;  k += stride) whhb[k]  = f2bf(whh[k]);
  for (int k = i; k < G4 * CDIM; k += stride) wihb[k]  = f2bf(wih[k]);
  for (int k = i; k < WDIM*COMB; k += stride) wlinb[k] = f2bf(wlin[k]);
  for (int k = i; k < G4;        k += stride) bcomb[k] = bih[k] + bhh[k];
}

// ------------------------------ main kernel --------------------------------
// 256 threads = 8 waves, 32 words per block.
// Wave w owns N-tiles {w, 16+w, 32+w, 48+w} of the 32x1024 gate GEMM.
__global__ __launch_bounds__(256, 1) void clenc_kernel(
    const int*   __restrict__ char_idx, const int* __restrict__ char_len,
    const float* __restrict__ word_emb, const float* __restrict__ E_char,
    const __bf16* __restrict__ wih,  const __bf16* __restrict__ whh,
    const __bf16* __restrict__ wlin, const float* __restrict__ bcomb,
    const float* __restrict__ blin,  float* __restrict__ out) {
  __shared__ __bf16 hbuf[ROWS * HID];    // 16 KB  (h state, bf16 A operand)
  __shared__ __bf16 xbuf[ROWS * CDIM];   //  4 KB  (x_t, bf16 A operand)
  __shared__ __bf16 comb[ROWS * COMB];   // 32 KB  (final [word_emb | h])

  const int tid  = threadIdx.x;
  const int lane = tid & 31;
  const int wave = tid >> 5;
  const int base = blockIdx.x * ROWS;
  const int lcol = lane & 15;            // column-within-tile
  const int rb   = (lane >> 4) * 8;      // row base within M-tile (C layout)
  const int hu   = wave * 16 + lcol;     // hidden unit owned by this lane

  // Zero h state (b128 stores: 8192 bf16 / 256 threads = 4 chunks of 8).
  {
    const uint4 z = {0u, 0u, 0u, 0u};
    #pragma unroll
    for (int it = 0; it < 4; ++it)
      *(uint4*)(hbuf + (tid + it * 256) * 8) = z;
  }

  // Warm L2 for the recurrent weights (global_prefetch_b8).
  for (int off = tid * 64; off < G4 * HID; off += 256 * 64)
    __builtin_prefetch(whh + off, 0, 1);

  // ---- register-resident weight fragments (loop-invariant, no spill) ------
  v16bf Bhh[4][8];                       // 4 gates x 8 K-tiles  (256 VGPRs)
  v16bf Bih[4][2];                       // 4 gates x 2 K-tiles  ( 64 VGPRs)
  #pragma unroll
  for (int j = 0; j < 4; ++j) {
    const int n0 = (j * 16 + wave) * 16 + lcol;   // gate j, hidden unit hu
    #pragma unroll
    for (int kt = 0; kt < 8; ++kt)
      Bhh[j][kt] = *(const v16bf*)(whh + n0 * HID + kt * 32 + (lane >> 4) * 16);
    #pragma unroll
    for (int kt = 0; kt < 2; ++kt)
      Bih[j][kt] = *(const v16bf*)(wih + n0 * CDIM + kt * 32 + (lane >> 4) * 16);
  }

  // Per-lane biases for the 4 gates of hidden unit hu.
  const float bi  = bcomb[hu];
  const float bf_ = bcomb[256 + hu];
  const float bg  = bcomb[512 + hu];
  const float bo  = bcomb[768 + hu];

  // Per-lane word lengths for the 16 rows this lane's C elements map to.
  int lenr[2][8];
  #pragma unroll
  for (int m = 0; m < 2; ++m)
    #pragma unroll
    for (int r = 0; r < 8; ++r)
      lenr[m][r] = char_len[base + m * 16 + rb + r];

  // c-state in registers: cst[m][r] for (row = m*16+rb+r, hidden unit hu).
  v8f cst[2];
  #pragma unroll
  for (int m = 0; m < 2; ++m)
    #pragma unroll
    for (int r = 0; r < 8; ++r) cst[m][r] = 0.0f;

  // Gather-phase coordinates: thread handles 8 contiguous embedding elems.
  const int grow = tid >> 3;             // 0..31
  const int ge0  = (tid & 7) * 8;        // 0,8,...,56

  __syncthreads();

  for (int t = 0; t < TSTEPS; ++t) {
    // -------- gather char embeddings: 1 idx + 2x b128 loads + 1x b128 store
    {
      const int ci = char_idx[(base + grow) * TSTEPS + t];
      const float4 f0 = *(const float4*)(E_char + ci * CDIM + ge0);
      const float4 f1 = *(const float4*)(E_char + ci * CDIM + ge0 + 4);
      uint4 pk;
      pk.x = pk2bf(f0.x, f0.y);
      pk.y = pk2bf(f0.z, f0.w);
      pk.z = pk2bf(f1.x, f1.y);
      pk.w = pk2bf(f1.z, f1.w);
      *(uint4*)(xbuf + grow * CDIM + ge0) = pk;
    }
    __syncthreads();

    // -------- gates = x_t @ W_ih^T + h @ W_hh^T, all operands resident -----
    v8f acc[2][4];
    #pragma unroll
    for (int m = 0; m < 2; ++m)
      #pragma unroll
      for (int j = 0; j < 4; ++j)
        #pragma unroll
        for (int r = 0; r < 8; ++r) acc[m][j][r] = 0.0f;

    #pragma unroll
    for (int kt = 0; kt < 2; ++kt) {                 // K = 64 over x_t
      const v16bf a0 = load_afrag(xbuf,             CDIM, kt, lane);
      const v16bf a1 = load_afrag(xbuf + 16 * CDIM, CDIM, kt, lane);
      #pragma unroll
      for (int j = 0; j < 4; ++j) {
        acc[0][j] = wmma_bf16(a0, Bih[j][kt], acc[0][j]);
        acc[1][j] = wmma_bf16(a1, Bih[j][kt], acc[1][j]);
      }
    }
    #pragma unroll
    for (int kt = 0; kt < 8; ++kt) {                 // K = 256 over h
      const v16bf a0 = load_afrag(hbuf,            HID, kt, lane);
      const v16bf a1 = load_afrag(hbuf + 16 * HID, HID, kt, lane);
      #pragma unroll
      for (int j = 0; j < 4; ++j) {
        acc[0][j] = wmma_bf16(a0, Bhh[j][kt], acc[0][j]);
        acc[1][j] = wmma_bf16(a1, Bhh[j][kt], acc[1][j]);
      }
    }
    __syncthreads();   // all waves done reading hbuf of step t

    // -------- LSTM cell entirely in registers on the C fragments -----------
    #pragma unroll
    for (int m = 0; m < 2; ++m) {
      #pragma unroll
      for (int r = 0; r < 8; ++r) {
        if (t < lenr[m][r]) {
          const float iv = sigmoidf_(acc[m][0][r] + bi);
          const float fv = sigmoidf_(acc[m][1][r] + bf_);
          const float gv = tanhf_   (acc[m][2][r] + bg);
          const float ov = sigmoidf_(acc[m][3][r] + bo);
          const float c  = fv * cst[m][r] + iv * gv;
          cst[m][r] = c;
          const int row = m * 16 + rb + r;
          hbuf[row * HID + hu] = f2bf(ov * tanhf_(c));
        }
      }
    }
    __syncthreads();   // new h visible (next iteration reads after this)
  }

  // -------- combined = [word_emb | h] in bf16, b128 staged -----------------
  #pragma unroll
  for (int it = 0; it < 4; ++it) {       // f32 word_emb -> bf16 (1024 chunks)
    const int u = tid + it * 256;
    const int row = u >> 5;              // 32 chunks of 8 per row
    const int c0  = (u & 31) * 8;
    const float4 f0 = *(const float4*)(word_emb + (size_t)(base + row) * WDIM + c0);
    const float4 f1 = *(const float4*)(word_emb + (size_t)(base + row) * WDIM + c0 + 4);
    uint4 pk;
    pk.x = pk2bf(f0.x, f0.y);
    pk.y = pk2bf(f0.z, f0.w);
    pk.z = pk2bf(f1.x, f1.y);
    pk.w = pk2bf(f1.z, f1.w);
    *(uint4*)(comb + row * COMB + c0) = pk;
  }
  #pragma unroll
  for (int it = 0; it < 4; ++it) {       // copy h (bf16) into second half
    const int u = tid + it * 256;
    const int row = u >> 5;
    const int c0  = (u & 31) * 8;
    *(uint4*)(comb + row * COMB + 256 + c0) = *(const uint4*)(hbuf + row * HID + c0);
  }
  __syncthreads();

  // -------- out = relu(combined @ W_lin^T + b_lin) -------------------------
  // Wave w owns output N-tiles {2w, 2w+1}; M = 2 tiles; K = 512 (16 k-tiles).
  v8f od[2][2];
  #pragma unroll
  for (int m = 0; m < 2; ++m)
    #pragma unroll
    for (int j = 0; j < 2; ++j)
      #pragma unroll
      for (int r = 0; r < 8; ++r) od[m][j][r] = 0.0f;

  #pragma unroll
  for (int kt = 0; kt < 16; ++kt) {
    const v16bf a0 = load_afrag(comb,             COMB, kt, lane);
    const v16bf a1 = load_afrag(comb + 16 * COMB, COMB, kt, lane);
    #pragma unroll
    for (int j = 0; j < 2; ++j) {
      const int n0 = (wave * 2 + j) * 16 + lcol;
      const v16bf b =
          *(const v16bf*)(wlin + n0 * COMB + kt * 32 + (lane >> 4) * 16);
      od[0][j] = wmma_bf16(a0, b, od[0][j]);
      od[1][j] = wmma_bf16(a1, b, od[1][j]);
    }
  }
  #pragma unroll
  for (int m = 0; m < 2; ++m) {
    #pragma unroll
    for (int j = 0; j < 2; ++j) {
      const int col = (wave * 2 + j) * 16 + lcol;
      const float bl = blin[col];
      #pragma unroll
      for (int r = 0; r < 8; ++r) {
        const float v = od[m][j][r] + bl;
        out[(size_t)(base + m * 16 + rb + r) * WDIM + col] = fmaxf(v, 0.0f);
      }
    }
  }
}

// ------------------------------- launcher ----------------------------------
extern "C" void kernel_launch(void* const* d_in, const int* in_sizes, int n_in,
                              void* d_out, int out_size, void* d_ws, size_t ws_size,
                              hipStream_t stream) {
  const int*   char_idx = (const int*)  d_in[0];
  const int*   char_len = (const int*)  d_in[1];
  const float* word_emb = (const float*)d_in[2];
  const float* E_char   = (const float*)d_in[3];
  const float* W_ih     = (const float*)d_in[4];
  const float* W_hh     = (const float*)d_in[5];
  const float* b_ih     = (const float*)d_in[6];
  const float* b_hh     = (const float*)d_in[7];
  const float* W_lin    = (const float*)d_in[8];
  const float* b_lin    = (const float*)d_in[9];
  float* out = (float*)d_out;

  const int N = in_sizes[2] / WDIM;   // 32768

  // Workspace carve (~900KB): bf16 weights + combined bias.
  char* ws = (char*)d_ws;
  __bf16* whh_b  = (__bf16*)(ws);                              // 512 KB
  __bf16* wih_b  = (__bf16*)(ws + 524288);                     // 128 KB
  __bf16* wlin_b = (__bf16*)(ws + 524288 + 131072);            // 256 KB
  float*  bcomb  = (float*) (ws + 524288 + 131072 + 262144);   //   4 KB

  clenc_pack_kernel<<<256, 256, 0, stream>>>(W_ih, W_hh, W_lin, b_ih, b_hh,
                                             wih_b, whh_b, wlin_b, bcomb);

  clenc_kernel<<<N / ROWS, 256, 0, stream>>>(
      char_idx, char_len, word_emb, E_char, wih_b, whh_b, wlin_b, bcomb,
      b_lin, out);
}
